// PairGraph_27127013442145
// MI455X (gfx1250) — matrix operand order repaired and checked by reading, not compiled, not run on hardware
//
#include <hip/hip_runtime.h>

// ---------------------------------------------------------------------------
// Types for CDNA5 WMMA (gfx1250, wave32).
// ---------------------------------------------------------------------------
typedef __bf16 bf16_t;
typedef __attribute__((ext_vector_type(16))) __bf16 v16bf;
typedef __attribute__((ext_vector_type(8)))  float  v8f;

#define B_SZ  64
#define C_SZ  512
#define HW_SZ 1024
#define IC_SZ 32

__device__ __forceinline__ v8f wmma_bf16(v16bf a, v16bf b, v8f c) {
    // D = A(16x32) * B(32x16) + C, fp32 accumulate
    return __builtin_amdgcn_wmma_f32_16x16x32_bf16(
        /*neg_a=*/false, a, /*neg_b=*/false, b,
        /*c_mod=*/(short)0, c, /*reuse_a=*/false, /*reuse_b=*/false);
}

// A-fragment (16x32 bf16, row-major source, leading dim `ld` elements).
// lane<16 -> row=lane, K=0..15; lane>=16 -> row=lane-16, K=16..31.
__device__ __forceinline__ v16bf load_a_frag(const bf16_t* base, int ld) {
    int lane = threadIdx.x & 31;
    const bf16_t* p = base + (size_t)(lane & 15) * ld + (lane >> 4) * 16;
    union { v16bf v; uint4 u[2]; } t;
    t.u[0] = *(const uint4*)p;          // K 0..7   (or 16..23)
    t.u[1] = *(const uint4*)(p + 8);    // K 8..15  (or 24..31)
    return t.v;
}

// B-fragment (32x16) from a [col][k] buffer (k contiguous, pitch ld elements).
// lane<16: col=lane, K=0..15 ; lane>=16: col=lane-16, K=16..31.
__device__ __forceinline__ v16bf load_b_frag_rowk(const bf16_t* base, int ld) {
    int lane = threadIdx.x & 31;
    const bf16_t* p = base + (size_t)(lane & 15) * ld + (lane >> 4) * 16;
    union { v16bf v; uint4 u[2]; } t;
    t.u[0] = *(const uint4*)p;
    t.u[1] = *(const uint4*)(p + 8);
    return t.v;
}

__device__ __forceinline__ unsigned pack_bf16(float a, float b) {
    union { unsigned u; bf16_t h[2]; } t;
    t.h[0] = (bf16_t)a; t.h[1] = (bf16_t)b;
    return t.u;
}

// ---------------------------------------------------------------------------
// k1: [Wy;Wg] (64x512) @ X[b] (512x1024) -> Y,G (each 32x1024).
//     Normalize Y columns (L2 over 32 channels), emit YhatT [b][j][m] (packed
//     b128 stores) and G [b][k][i] in bf16.
// grid (8 jtiles, 64 batches), block 256 (8 waves). Tile: 64 x 128.
// ---------------------------------------------------------------------------
__global__ void k1_yg(const float* __restrict__ x, const float* __restrict__ Wy,
                      const float* __restrict__ Wg,
                      bf16_t* __restrict__ YhatT, bf16_t* __restrict__ Gbuf) {
    __shared__ bf16_t Ws[64 * 32];    // [m][k]  4 KB
    __shared__ bf16_t Xs[128 * 32];   // [j][k]  8 KB (transposed for B-frags)
    const int b    = blockIdx.y;
    const int j0   = blockIdx.x * 128;
    const int tid  = threadIdx.x;
    const int wave = tid >> 5;
    const int lane = tid & 31;
    const float* xb = x + (size_t)b * C_SZ * HW_SZ;

    v8f acc[4] = {v8f{}, v8f{}, v8f{}, v8f{}};

    for (int k0 = 0; k0 < C_SZ; k0 += 32) {
#pragma unroll
        for (int e = 0; e < 4; ++e) {            // stage [Wy;Wg] 64x32 as pairs
            int idx = tid + e * 256;             // pair index: m*16 + kp
            int m = idx >> 4, kp = idx & 15;
            const float* src = (m < 32) ? (Wy + (size_t)m * C_SZ)
                                        : (Wg + (size_t)(m - 32) * C_SZ);
            float2 w = *(const float2*)(src + k0 + 2 * kp);
            ((unsigned*)Ws)[idx] = pack_bf16(w.x, w.y);
        }
#pragma unroll
        for (int e = 0; e < 8; ++e) {            // stage X 32x128 -> [j][k] pairs
            int idx = tid + e * 256;             // 0..2047
            int kp = idx >> 7, jj = idx & 127;
            float w0 = xb[(size_t)(k0 + 2 * kp) * HW_SZ + j0 + jj];
            float w1 = xb[(size_t)(k0 + 2 * kp + 1) * HW_SZ + j0 + jj];
            ((unsigned*)Xs)[jj * 16 + kp] = pack_bf16(w0, w1);
        }
        if (k0 + 32 < C_SZ)                      // global_prefetch_b8 next tile
            __builtin_prefetch(xb + (size_t)(k0 + 32) * HW_SZ + j0 + (tid & 127), 0, 1);
        __syncthreads();
        v16bf bx = load_b_frag_rowk(&Xs[wave * 16 * 32], 32);
#pragma unroll
        for (int ms = 0; ms < 4; ++ms) {
            v16bf aw = load_a_frag(&Ws[ms * 16 * 32], 32);
            acc[ms] = wmma_bf16(aw, bx, acc[ms]);
        }
        __syncthreads();
    }

    // Column L2 norm over the 32 Y rows (frags 0,1); lane^16 holds the
    // complementary 16 rows of the same column.
    float ss = 0.f;
#pragma unroll
    for (int r = 0; r < 8; ++r) ss += acc[0][r] * acc[0][r] + acc[1][r] * acc[1][r];
    ss += __shfl_xor(ss, 16, 32);
    float scale = 1.0f / (sqrtf(ss) + 1e-12f);

    const int col   = j0 + wave * 16 + (lane & 15);
    const int rbase = (lane >> 4) * 8;
    const size_t ytbase = ((size_t)b * HW_SZ + col) * IC_SZ;
#pragma unroll
    for (int ms = 0; ms < 2; ++ms) {             // Yhat^T: contiguous in m
        union { uint4 u; bf16_t h[8]; } pk;
#pragma unroll
        for (int r = 0; r < 8; ++r) pk.h[r] = (bf16_t)(acc[ms][r] * scale);
        *(uint4*)(YhatT + ytbase + ms * 16 + rbase) = pk.u;
    }
#pragma unroll
    for (int ms = 2; ms < 4; ++ms) {             // G row-major [k][i]
#pragma unroll
        for (int r = 0; r < 8; ++r) {
            int k = (ms - 2) * 16 + rbase + r;
            Gbuf[((size_t)b * IC_SZ + k) * HW_SZ + col] = (bf16_t)acc[ms][r];
        }
    }
}

// ---------------------------------------------------------------------------
// k2: S_j = sum_i relu(yhat_i . yhat_j) (full column sum), dnorm_j.
// grid (8, 64), block 256; wave w owns 16 columns; K-loop over all 1024 i.
// Both operands come from YhatT via vectorized fragment loads.
// ---------------------------------------------------------------------------
__global__ void k2_deg(const bf16_t* __restrict__ YhatT,
                       float* __restrict__ S, float* __restrict__ Dn) {
    const int b = blockIdx.y, j0 = blockIdx.x * 128;
    const int tid = threadIdx.x, wave = tid >> 5, lane = tid & 31;
    const bf16_t* ytb = YhatT + (size_t)b * HW_SZ * IC_SZ;
    const int jcol0 = j0 + wave * 16;

    v16bf by = load_b_frag_rowk(ytb + (size_t)jcol0 * IC_SZ, IC_SZ);  // fixed B
    float sacc = 0.f;
    for (int i0 = 0; i0 < HW_SZ; i0 += 16) {
        v16bf ay = load_a_frag(ytb + (size_t)i0 * IC_SZ, IC_SZ);
        v8f c = {};
        c = wmma_bf16(ay, by, c);
#pragma unroll
        for (int r = 0; r < 8; ++r) sacc += fmaxf(c[r], 0.f);
    }
    sacc += __shfl_xor(sacc, 16, 32);
    float cjj = 0.f;                       // ||yhat_j||^2 from the B fragment
#pragma unroll
    for (int e = 0; e < 16; ++e) { float v = (float)by[e]; cjj += v * v; }
    cjj += __shfl_xor(cjj, 16, 32);
    if (lane < 16) {
        int j = jcol0 + lane;
        float deg = sacc - fmaxf(cjj, 0.f);     // remove diagonal
        S [(size_t)b * HW_SZ + j] = sacc;
        Dn[(size_t)b * HW_SZ + j] = rsqrtf(deg > 1e-3f ? deg : 1.0f);
    }
}

// ---------------------------------------------------------------------------
// k3: fused T' = G @ (diag(dn) * relu(Yhat^T Yhat)), then
//     o[k,j] = dn_j*(g[k,j]*dn_j*S_j - T'[k,j]) written transposed [b][j][k].
// Per 32-wide i-chunk: 2 WMMA produce C (32 x 16cols) -> relu*dn_i -> LDS
// [j][i] -> 2 WMMA accumulate T'. Similarity matrix never touches HBM; the
// dn row-scaling is folded in, so no separate GH pass/buffer is needed.
// grid (8, 64), block 256.
// ---------------------------------------------------------------------------
__global__ void k3_T(const bf16_t* __restrict__ YhatT, const bf16_t* __restrict__ Gbuf,
                     const float* __restrict__ S, const float* __restrict__ Dn,
                     bf16_t* __restrict__ ObufT) {
    __shared__ bf16_t cbuf[128 * 32];   // [j][i] per-wave 16-col slices, 8 KB
    const int b = blockIdx.y, j0 = blockIdx.x * 128;
    const int tid = threadIdx.x, wave = tid >> 5, lane = tid & 31;
    const bf16_t* ytb = YhatT + (size_t)b * HW_SZ * IC_SZ;
    const bf16_t* gb  = Gbuf  + (size_t)b * IC_SZ * HW_SZ;
    const float*  Dnb = Dn    + (size_t)b * HW_SZ;
    const int jcol0 = j0 + wave * 16;
    const int rbase = (lane >> 4) * 8;

    v16bf by = load_b_frag_rowk(ytb + (size_t)jcol0 * IC_SZ, IC_SZ); // fixed
    v8f t0 = {}, t1 = {};
    bf16_t* cw = cbuf + wave * 16 * 32;                     // this wave's cols

    for (int i0 = 0; i0 < HW_SZ; i0 += 32) {
        __syncthreads();
#pragma unroll
        for (int s = 0; s < 2; ++s) {
            v16bf ay = load_a_frag(ytb + (size_t)(i0 + s * 16) * IC_SZ, IC_SZ);
            v8f c = {};
            c = wmma_bf16(ay, by, c);                       // C tile (i x j)
            // dn for this lane's 8 contiguous rows i = i0+s*16+rbase+r
            float4 d0 = *(const float4*)(Dnb + i0 + s * 16 + rbase);
            float4 d1 = *(const float4*)(Dnb + i0 + s * 16 + rbase + 4);
            float dnv[8] = {d0.x, d0.y, d0.z, d0.w, d1.x, d1.y, d1.z, d1.w};
            union { uint4 u; bf16_t h[8]; } pk;
#pragma unroll
            for (int r = 0; r < 8; ++r)
                pk.h[r] = (bf16_t)(fmaxf(c[r], 0.f) * dnv[r]);
            // D-frag row i = s*16 + rbase + r, col j = lane&15 -> [j][i]
            *(uint4*)(cw + (size_t)(lane & 15) * 32 + s * 16 + rbase) = pk.u;
        }
        __syncthreads();
        v16bf bc = load_b_frag_rowk(cw, 32);                // dn*relu(C) as B
        v16bf g0 = load_a_frag(gb + i0, HW_SZ);             // G rows 0..15
        v16bf g1 = load_a_frag(gb + (size_t)16 * HW_SZ + i0, HW_SZ);
        t0 = wmma_bf16(g0, bc, t0);
        t1 = wmma_bf16(g1, bc, t1);
    }

    const int j = jcol0 + (lane & 15);
    const float sj = S [(size_t)b * HW_SZ + j];
    const float dj = Dnb[j];
    const size_t otbase = ((size_t)b * HW_SZ + j) * IC_SZ;
#pragma unroll
    for (int h = 0; h < 2; ++h) {
        v8f t = h ? t1 : t0;
        union { uint4 u; bf16_t h8[8]; } pk;
#pragma unroll
        for (int r = 0; r < 8; ++r) {
            int k = h * 16 + rbase + r;
            float g = (float)gb[(size_t)k * HW_SZ + j];
            pk.h8[r] = (bf16_t)(dj * (g * dj * sj - t[r]));
        }
        *(uint4*)(ObufT + otbase + h * 16 + rbase) = pk.u;  // contiguous in k
    }
}

// ---------------------------------------------------------------------------
// k4: out = x + Wo (512x32) @ o (32x1024). o comes from ObufT [b][j][k], so
// B-fragments load straight from global with two b128 loads (no LDS staging).
// grid (8 jtiles, 8 mtiles, 64 b).
// ---------------------------------------------------------------------------
__global__ void k4_out(const float* __restrict__ x, const float* __restrict__ Wo,
                       const bf16_t* __restrict__ ObufT, float* __restrict__ out) {
    __shared__ bf16_t Wos[64 * 32];   // [m][k]
    const int b = blockIdx.z, m0 = blockIdx.y * 64, j0 = blockIdx.x * 128;
    const int tid = threadIdx.x, wave = tid >> 5, lane = tid & 31;

#pragma unroll
    for (int e = 0; e < 4; ++e) {                // stage Wo tile as bf16 pairs
        int idx = tid + e * 256;                 // pair index m*16 + kp
        int m = idx >> 4, kp = idx & 15;
        float2 w = *(const float2*)(Wo + (size_t)(m0 + m) * IC_SZ + 2 * kp);
        ((unsigned*)Wos)[idx] = pack_bf16(w.x, w.y);
    }
    __syncthreads();

    const bf16_t* obT = ObufT + (size_t)b * HW_SZ * IC_SZ;
    v16bf bo = load_b_frag_rowk(obT + (size_t)(j0 + wave * 16) * IC_SZ, IC_SZ);
    const float* xb   = x   + (size_t)b * C_SZ * HW_SZ;
    float*       outb = out + (size_t)b * C_SZ * HW_SZ;
    const int j = j0 + wave * 16 + (lane & 15);
    const int rbase = (lane >> 4) * 8;
#pragma unroll
    for (int ms = 0; ms < 4; ++ms) {
        v16bf aw = load_a_frag(&Wos[ms * 16 * 32], 32);
        v8f acc = {};
        acc = wmma_bf16(aw, bo, acc);
#pragma unroll
        for (int r = 0; r < 8; ++r) {
            int m = m0 + ms * 16 + rbase + r;
            outb[(size_t)m * HW_SZ + j] = xb[(size_t)m * HW_SZ + j] + acc[r];
        }
    }
}

// ---------------------------------------------------------------------------
// Host launcher. Workspace layout (12.5 MB, bf16 intermediates live in L2):
//   [0,4M)       YhatT  bf16 [B][1024][32]
//   [4M,8M)      Gbuf   bf16 [B][32][1024]
//   [8M..]       S f32 [B][1024], Dn f32 [B][1024]
//   [9M,13M)     ObufT  bf16 [B][1024][32]
// ---------------------------------------------------------------------------
extern "C" void kernel_launch(void* const* d_in, const int* in_sizes, int n_in,
                              void* d_out, int out_size, void* d_ws, size_t ws_size,
                              hipStream_t stream) {
    (void)in_sizes; (void)n_in; (void)out_size; (void)ws_size;
    const float* x  = (const float*)d_in[0];
    const float* Wy = (const float*)d_in[1];
    const float* Wg = (const float*)d_in[2];
    const float* Wo = (const float*)d_in[3];
    float* out = (float*)d_out;

    char* ws = (char*)d_ws;
    bf16_t* YhatT = (bf16_t*)(ws);
    bf16_t* Gbuf  = (bf16_t*)(ws + (4u << 20));
    float*  S     = (float*) (ws + (8u << 20));
    float*  Dn    = (float*) (ws + (8u << 20) + (256u << 10));
    bf16_t* ObufT = (bf16_t*)(ws + (9u << 20));

    k1_yg <<<dim3(8, 64),    256, 0, stream>>>(x, Wy, Wg, YhatT, Gbuf);
    k2_deg<<<dim3(8, 64),    256, 0, stream>>>(YhatT, S, Dn);
    k3_T  <<<dim3(8, 64),    256, 0, stream>>>(YhatT, Gbuf, S, Dn, ObufT);
    k4_out<<<dim3(8, 8, 64), 256, 0, stream>>>(x, Wo, ObufT, out);
}
